// MoEGenreGate_13357348291295
// MI455X (gfx1250) — matrix operand order repaired and checked by reading, not compiled, not run on hardware
//
#include <hip/hip_runtime.h>

typedef __attribute__((ext_vector_type(16))) __bf16 v16bf;
typedef __attribute__((ext_vector_type(8)))  float  v8f;

#define Bsz   4
#define Ssz   8192
#define Dsz   1024
#define DGsz  256
#define Esz   64
#define NTOK  (Bsz * Ssz)        // 32768 tokens
#define KITER (Dsz / 32)         // 32 K-steps of 32

// ---------------------------------------------------------------------------
// Pack Wx (D x E, fp32 row-major) into bf16 WMMA B-fragment order.
// B-matrix 32x16 bf16 order used here (contiguous K, same as A): lane n in 0..15 holds column N=n,
// K = 0..15 packed 2-per-dword; lanes 16..31 hold K = 16..31.
// Bpack[kk][t][lane][j]: j = 0..15 bf16 -> K = kk*32 + (lane&16) + j.
// ---------------------------------------------------------------------------
__global__ __launch_bounds__(128)
void pack_Wx_kernel(const float* __restrict__ Wx, __bf16* __restrict__ Bpack) {
    int tid  = blockIdx.x * 128 + threadIdx.x;   // 4096 = 32 kk * 4 t * 32 lane
    int kk   = tid >> 7;
    int t    = (tid >> 5) & 3;
    int lane = tid & 31;
    int e     = t * 16 + (lane & 15);
    int kbase = kk * 32 + ((lane & 16) ? 16 : 0);
    __bf16* dst = Bpack + (size_t)tid * 16;      // tid == ((kk*4+t)*32+lane)
#pragma unroll
    for (int j = 0; j < 16; ++j)
        dst[j] = (__bf16)Wx[(size_t)(kbase + j) * Esz + e];
}

// ---------------------------------------------------------------------------
// gbias[b][e] = genre_emb[b] @ Wg[:,e] + bias[e]   (only 256 outputs)
// ---------------------------------------------------------------------------
__global__ __launch_bounds__(256)
void genre_bias_kernel(const float* __restrict__ ge, const float* __restrict__ Wg,
                       const float* __restrict__ bias, float* __restrict__ gbias) {
    int tid = threadIdx.x;           // 256 = 4 batches * 64 experts
    int b = tid >> 6, e = tid & 63;
    float s = bias[e];
    for (int k = 0; k < DGsz; ++k)
        s += ge[b * DGsz + k] * Wg[k * Esz + e];
    gbias[tid] = s;
}

// ---------------------------------------------------------------------------
// Main fused router: per wave, 16 tokens x 64 experts.
// ---------------------------------------------------------------------------
__global__ __launch_bounds__(256)
void moe_router_kernel(const float* __restrict__ x,
                       const __bf16* __restrict__ Bpack,
                       const float* __restrict__ gbias,
                       float* __restrict__ gates,
                       int* __restrict__ topk_idx) {
    __shared__ float lds[8 * 16 * 64];           // 32 KB: 8 waves x 16 tok x 64 exp

    const int wave   = threadIdx.x >> 5;
    const int lane   = threadIdx.x & 31;
    const int gwave  = blockIdx.x * 8 + wave;
    const int token0 = gwave * 16;
    const int batch  = token0 >> 13;             // / 8192 (16 | 8192, no straddle)

    // A-fragment addressing (matches Bpack): lane L<16 owns row M=L,
    // K = 0..15; lane L+16 owns row M=L, K = 16..31.
    const int    row   = token0 + (lane & 15);
    const int    half8 = (lane & 16) ? 16 : 0;
    const float* xrow  = x + (size_t)row * Dsz + half8;
    const __bf16* bln  = Bpack + (size_t)lane * 16;

    const v8f vzero = {0.f, 0.f, 0.f, 0.f, 0.f, 0.f, 0.f, 0.f};
    v8f acc[4] = {vzero, vzero, vzero, vzero};

    for (int kk = 0; kk < KITER; ++kk) {
        const float* p = xrow + kk * 32;
        __builtin_prefetch(p + 128, 0, 1);       // stream hint, 4 K-steps ahead
        float4 f0 = *(const float4*)(p);
        float4 f1 = *(const float4*)(p + 4);
        float4 f2 = *(const float4*)(p + 8);
        float4 f3 = *(const float4*)(p + 12);
        v16bf a;
        a[0]  = (__bf16)f0.x; a[1]  = (__bf16)f0.y; a[2]  = (__bf16)f0.z; a[3]  = (__bf16)f0.w;
        a[4]  = (__bf16)f1.x; a[5]  = (__bf16)f1.y; a[6]  = (__bf16)f1.z; a[7]  = (__bf16)f1.w;
        a[8]  = (__bf16)f2.x; a[9]  = (__bf16)f2.y; a[10] = (__bf16)f2.z; a[11] = (__bf16)f2.w;
        a[12] = (__bf16)f3.x; a[13] = (__bf16)f3.y; a[14] = (__bf16)f3.z; a[15] = (__bf16)f3.w;

        const __bf16* bk = bln + (size_t)kk * (4 * 32 * 16);
#pragma unroll
        for (int t = 0; t < 4; ++t) {
            v16bf bfrag = *(const v16bf*)(bk + (size_t)t * (32 * 16));
            acc[t] = __builtin_amdgcn_wmma_f32_16x16x32_bf16(
                false, a, false, bfrag, (short)0, acc[t], false, false);
        }
    }

    // Scatter C fragments (16x16 f32: VGPR r, lanes 0..15 -> M=r, lanes 16..31 -> M=r+8)
    float* myl = lds + wave * (16 * 64);
    const int nCol = lane & 15;
    const int mOff = (lane >> 4) << 3;
#pragma unroll
    for (int t = 0; t < 4; ++t)
#pragma unroll
        for (int r = 0; r < 8; ++r)
            myl[(mOff + r) * 64 + t * 16 + nCol] = acc[t][r];
    __syncthreads();

    // Per-token top-2 + renormalized gates. Softmax denominator cancels:
    // g1 = 1/(1+exp(l2-l1)), g2 = 1-g1.
    if (lane < 16) {
        const float* lrow = myl + lane * 64;
        const float* gb   = gbias + batch * 64;
        float l1 = -3.4e38f, l2 = -3.4e38f;
        int   i1 = 0,        i2 = 0;
        for (int e = 0; e < 64; ++e) {
            float v = lrow[e] + gb[e];
            if (v > l1)      { l2 = l1; i2 = i1; l1 = v; i1 = e; }
            else if (v > l2) { l2 = v;  i2 = e; }
        }
        float rr = __expf(l2 - l1);
        float g1 = 1.0f / (1.0f + rr);
        float g2 = rr * g1;

        const int token = token0 + lane;
        float* grow = gates + (size_t)token * 64;
        float4 z = make_float4(0.f, 0.f, 0.f, 0.f);
#pragma unroll
        for (int q = 0; q < 16; ++q)
            *(float4*)(grow + q * 4) = z;
        grow[i1] = g1;                   // same-wave store order to same channel is kept
        grow[i2] = g2;
        topk_idx[token * 2 + 0] = i1;
        topk_idx[token * 2 + 1] = i2;
    }
}

// ---------------------------------------------------------------------------
// Inputs (setup_inputs order): x[B,S,D] f32, genre_emb[B,DG] f32,
// Wx[D,E] f32, Wg[DG,E] f32, b[E] f32.
// Outputs: gates [B,S,E] f32 (2097152), then topk_idx [B,S,K] i32 (65536).
// ---------------------------------------------------------------------------
extern "C" void kernel_launch(void* const* d_in, const int* in_sizes, int n_in,
                              void* d_out, int out_size, void* d_ws, size_t ws_size,
                              hipStream_t stream) {
    const float* x     = (const float*)d_in[0];
    const float* genre = (const float*)d_in[1];
    const float* Wx    = (const float*)d_in[2];
    const float* Wg    = (const float*)d_in[3];
    const float* bias  = (const float*)d_in[4];

    __bf16* Bpack = (__bf16*)d_ws;                                   // 131072 B
    float*  gbias = (float*)((char*)d_ws + (size_t)KITER * 4 * 32 * 16 * sizeof(__bf16));

    float* gates   = (float*)d_out;
    int*   topkidx = (int*)(gates + (size_t)NTOK * Esz);

    pack_Wx_kernel<<<32, 128, 0, stream>>>(Wx, Bpack);
    genre_bias_kernel<<<1, 256, 0, stream>>>(genre, Wg, bias, gbias);
    moe_router_kernel<<<NTOK / (16 * 8), 256, 0, stream>>>(x, Bpack, gbias, gates, topkidx);
}
